// TreeDecoderTeacherForced_16458314678345
// MI455X (gfx1250) — compile-verified, matrix operand order checked by reading, not run on hardware
//
#include <hip/hip_runtime.h>
#include <hip/hip_bf16.h>
#include <stdint.h>

// ---------------------------------------------------------------------------
// Types for CDNA5 WMMA (wave32, 16x16x32 bf16 -> f32)
// ---------------------------------------------------------------------------
typedef __attribute__((ext_vector_type(16))) __bf16 v16bf;
typedef __attribute__((ext_vector_type(8)))  float  v8f;

union Frag16 {            // 16 bf16 = 32 bytes = two 16B chunks
    uint4 q[2];
    v16bf v;
};

__device__ __forceinline__ unsigned short f2bf(float f) {
    unsigned int u = __float_as_uint(f);
    unsigned int r = u + 0x7FFFu + ((u >> 16) & 1u);   // round-to-nearest-even
    return (unsigned short)(r >> 16);
}
__device__ __forceinline__ float bf2f(unsigned short b) {
    return __uint_as_float(((unsigned int)b) << 16);
}

#define HID        64
#define ROWS_BLK   64    // rows per block (2 row-groups x 2 tiles of 16)
#define AP         584   // LDS row stride for gathered A (576 + 8 pad), bf16 units
#define HP         72    // LDS row stride for H tile (64 + 8 pad), bf16 units
#define LDS_BYTES  ((ROWS_BLK * AP + ROWS_BLK * HP) * 2)   // 83,968 B

// ---------------------------------------------------------------------------
// f32 -> bf16 bulk convert (4 elems / thread)
// ---------------------------------------------------------------------------
__global__ void k_cvt_bf16(const float* __restrict__ in,
                           unsigned short* __restrict__ out, int n4) {
    int t = blockIdx.x * blockDim.x + threadIdx.x;
    if (t >= n4) return;
    const float4 f = *(const float4*)(in + (size_t)t * 4);
    unsigned int lo = (unsigned int)f2bf(f.x) | ((unsigned int)f2bf(f.y) << 16);
    unsigned int hi = (unsigned int)f2bf(f.z) | ((unsigned int)f2bf(f.w) << 16);
    uint2 v; v.x = lo; v.y = hi;
    *(uint2*)(out + (size_t)t * 4) = v;
}

// Zero the pad rows used to absorb -1 neighbors / tail rows.
__global__ void k_zero_pads(unsigned short* __restrict__ p0,
                            unsigned short* __restrict__ p1) {
    int t = threadIdx.x;                 // 16 threads, 16B each
    uint4 z = make_uint4(0u, 0u, 0u, 0u);
    if (t < 8)  *(uint4*)(p0 + t * 8) = z;
    else        *(uint4*)(p1 + (t - 8) * 8) = z;
}

// ---------------------------------------------------------------------------
// Pack weights [K, 64] f32 row-major into per-lane B fragments:
//   Wp[((kb*4 + nb)*32 + lane)*16 + i], lane holds col = nb*16 + lane%16,
//   elems 0..7 -> K kb*32 + (lane/16)*8 + 0..7 ; elems 8..15 -> +16.
// ---------------------------------------------------------------------------
__global__ void k_pack_w(const float* __restrict__ W,
                         unsigned short* __restrict__ Wp, int K) {
    int t = blockIdx.x * blockDim.x + threadIdx.x;
    int total = K * 64;
    if (t >= total) return;
    int i    = t & 15;
    int lane = (t >> 4) & 31;
    int nb   = (t >> 9) & 3;
    int kb   = t >> 11;
    int col  = nb * 16 + (lane & 15);
    int kloc = ((i < 8) ? i : (i + 8)) + ((lane >> 4) << 3);
    int k    = kb * 32 + kloc;
    Wp[t] = f2bf(W[(size_t)k * 64 + col]);
}

// ---------------------------------------------------------------------------
// child_in[i] = h_bf16[parent[i]] * gate(split_gt[parent[i]])  (gate is 0/1)
// ---------------------------------------------------------------------------
__global__ void k_child_gather(const unsigned short* __restrict__ h,
                               const long long* __restrict__ parent,
                               const float* __restrict__ gate,
                               unsigned short* __restrict__ child, int m8) {
    int t = blockIdx.x * blockDim.x + threadIdx.x;
    if (t >= m8) return;
    int i    = t >> 3;
    int part = t & 7;
    long long p = parent[i];
    float g = gate[p];
    uint4 v = make_uint4(0u, 0u, 0u, 0u);
    if (g != 0.0f)
        v = *(const uint4*)(h + (size_t)p * HID + part * 8);
    *(uint4*)(child + (size_t)i * HID + part * 8) = v;
}

// ---------------------------------------------------------------------------
// Fused QuadConv layer:  h = relu(gather9(in) @ W + b)
//                        split = h @ Ws ; value = h @ Wv
// Block: 256 threads (8 waves) covers 64 rows.
// wave -> (rowgroup = wave>>2 in 0..1, col = wave&3); each wave computes two
// 16-row tiles sharing every B fragment (halves weight traffic from L2).
// Gather uses GLOBAL_LOAD_ASYNC_TO_LDS_B128 (ASYNCcnt path, L2 -> LDS direct);
// -1 neighbors and tail rows route to a zeroed pad row (index pad_row).
// ---------------------------------------------------------------------------
__global__ void __launch_bounds__(256)
k_quadconv(const unsigned short* __restrict__ in_bf,     // [*, 64] bf16 feats
           const long long* __restrict__ neigh,          // [nrows, 9]
           const unsigned short* __restrict__ Wp,        // packed [18][4][32][16]
           const float* __restrict__ bias,               // [64]
           const unsigned short* __restrict__ Wvp,       // packed [2][4][32][16]
           const float* __restrict__ Ws,                 // [64]
           unsigned short* __restrict__ h_out,           // [nrows,64] bf16 or null
           float* __restrict__ split_out,                // [nrows]
           float* __restrict__ value_out,                // [nrows,64]
           int nrows, int pad_row)
{
    extern __shared__ unsigned short smem[];
    unsigned short* Abuf = smem;                       // [64 rows][AP] bf16
    unsigned short* Hbuf = smem + ROWS_BLK * AP;       // [64 rows][HP] bf16

    const int tid     = threadIdx.x;
    const int rowbase = blockIdx.x * ROWS_BLK;

    // ---- stage 1: async gather 64 rows x 9 neighbors x 64 ch into LDS -----
    // 64*9 = 576 segments of 128B, each split into 8 x 16B per-lane copies.
    // 576*8 / 256 = 18 iterations exactly -> EXEC stays all-ones.
    #pragma unroll 3
    for (int t = tid; t < ROWS_BLK * 9 * 8; t += 256) {
        int seg  = t >> 3;
        int part = t & 7;
        int row  = seg / 9;
        int j    = seg - row * 9;
        int grow = rowbase + row;
        long long g = (grow < nrows) ? neigh[(long long)grow * 9 + j] : -1ll;
        long long srow = (g >= 0) ? g : (long long)pad_row;
        const unsigned short* src = in_bf + (size_t)srow * HID + part * 8;
        unsigned lds = (unsigned)(uintptr_t)(&Abuf[row * AP + j * 64 + part * 8]);
        asm volatile("global_load_async_to_lds_b128 %0, %1, off"
                     :: "v"(lds), "v"(src) : "memory");
    }
    asm volatile("s_wait_asynccnt 0x0" ::: "memory");
    __syncthreads();

    // ---- stage 2: main GEMM, two 16-row tiles per wave share each B frag ---
    const int lane     = tid & 31;
    const int wave     = tid >> 5;
    const int rowgroup = wave >> 2;          // 0..1  (32 rows each)
    const int col      = wave & 3;           // 0..3  (16 cols each)
    const int laneRow  = lane & 15;
    const int laneHi   = lane >> 4;          // 0 or 1
    const int Arow0    = rowgroup * 32 + laneRow;      // tile 0
    const int Arow1    = Arow0 + 16;                   // tile 1

    v8f acc0 = {}, acc1 = {};
    #pragma unroll
    for (int kb = 0; kb < 18; ++kb) {
        Frag16 b, a0, a1;
        const unsigned short* bp = Wp + ((size_t)(kb * 4 + col) * 32 + lane) * 16;
        b.q[0] = *(const uint4*)(bp);
        b.q[1] = *(const uint4*)(bp + 8);
        const unsigned short* ap0 = &Abuf[Arow0 * AP + kb * 32 + laneHi * 8];
        a0.q[0] = *(const uint4*)(ap0);
        a0.q[1] = *(const uint4*)(ap0 + 16);
        const unsigned short* ap1 = &Abuf[Arow1 * AP + kb * 32 + laneHi * 8];
        a1.q[0] = *(const uint4*)(ap1);
        a1.q[1] = *(const uint4*)(ap1 + 16);
        acc0 = __builtin_amdgcn_wmma_f32_16x16x32_bf16(
                   false, a0.v, false, b.v, (short)0, acc0, false, false);
        acc1 = __builtin_amdgcn_wmma_f32_16x16x32_bf16(
                   false, a1.v, false, b.v, (short)0, acc1, false, false);
    }

    // ---- epilogue: bias + relu, spill h to global(bf16) + LDS --------------
    const int n  = col * 16 + laneRow;
    const float bv = bias[n];
    #pragma unroll
    for (int r = 0; r < 8; ++r) {
        #pragma unroll
        for (int tile = 0; tile < 2; ++tile) {
            float v = ((tile == 0) ? acc0[r] : acc1[r]) + bv;
            v = v > 0.0f ? v : 0.0f;
            unsigned short hb = f2bf(v);
            int lrow = rowgroup * 32 + tile * 16 + r + laneHi * 8;
            Hbuf[lrow * HP + n] = hb;
            int grow = rowbase + lrow;
            if (h_out && grow < nrows)
                h_out[(size_t)grow * HID + n] = hb;
        }
    }
    __syncthreads();

    // ---- stage 3: value head, B fragments again shared across both tiles ---
    v8f vac0 = {}, vac1 = {};
    #pragma unroll
    for (int kb = 0; kb < 2; ++kb) {
        Frag16 b, a0, a1;
        const unsigned short* bp = Wvp + ((size_t)(kb * 4 + col) * 32 + lane) * 16;
        b.q[0] = *(const uint4*)(bp);
        b.q[1] = *(const uint4*)(bp + 8);
        const unsigned short* hp0 = &Hbuf[Arow0 * HP + kb * 32 + laneHi * 8];
        a0.q[0] = *(const uint4*)(hp0);
        a0.q[1] = *(const uint4*)(hp0 + 16);
        const unsigned short* hp1 = &Hbuf[Arow1 * HP + kb * 32 + laneHi * 8];
        a1.q[0] = *(const uint4*)(hp1);
        a1.q[1] = *(const uint4*)(hp1 + 16);
        vac0 = __builtin_amdgcn_wmma_f32_16x16x32_bf16(
                   false, a0.v, false, b.v, (short)0, vac0, false, false);
        vac1 = __builtin_amdgcn_wmma_f32_16x16x32_bf16(
                   false, a1.v, false, b.v, (short)0, vac1, false, false);
    }
    #pragma unroll
    for (int r = 0; r < 8; ++r) {
        #pragma unroll
        for (int tile = 0; tile < 2; ++tile) {
            int lrow = rowgroup * 32 + tile * 16 + r + laneHi * 8;
            int grow = rowbase + lrow;
            if (grow < nrows)
                value_out[(size_t)grow * HID + n] =
                    (tile == 0) ? vac0[r] : vac1[r];
        }
    }

    // ---- stage 4: split head (VALU dot, 1 row per lane of waves 0-1) -------
    if (tid < ROWS_BLK) {
        int grow = rowbase + tid;
        if (grow < nrows) {
            float s = 0.0f;
            #pragma unroll 8
            for (int k = 0; k < HID; ++k)
                s += bf2f(Hbuf[tid * HP + k]) * Ws[k];
            split_out[grow] = s;
        }
    }
}

// ---------------------------------------------------------------------------
// Host-side launch
// ---------------------------------------------------------------------------
extern "C" void kernel_launch(void* const* d_in, const int* in_sizes, int n_in,
                              void* d_out, int out_size, void* d_ws, size_t ws_size,
                              hipStream_t stream) {
    (void)in_sizes; (void)n_in; (void)out_size; (void)ws_size;

    const int NP = 200000;          // parents
    const int NC = 300000;          // children

    const float*     feat    = (const float*)d_in[0];
    const long long* neigh_d = (const long long*)d_in[1];
    const long long* neigh_c = (const long long*)d_in[2];
    const long long* parent  = (const long long*)d_in[3];
    const float*     gate    = (const float*)d_in[4];
    const float*     W1      = (const float*)d_in[5];
    const float*     b1      = (const float*)d_in[6];
    const float*     W2      = (const float*)d_in[7];
    const float*     b2      = (const float*)d_in[8];
    const float*     Ws1     = (const float*)d_in[9];
    const float*     Wv1     = (const float*)d_in[10];
    const float*     Ws2     = (const float*)d_in[11];
    const float*     Wv2     = (const float*)d_in[12];

    float* out = (float*)d_out;
    float* split1 = out;
    float* value1 = out + NP;
    float* split2 = out + NP + (size_t)NP * HID;           // 13,000,000
    float* value2 = split2 + NC;                           // 13,300,000

    // workspace carve-up (bytes); featb/childb carry one extra zero pad row
    char* ws = (char*)d_ws;
    unsigned short* featb  = (unsigned short*)(ws);                 // (NP+1)*128 B
    unsigned short* h1b    = (unsigned short*)(ws + 25600256);      // NP*128 B
    unsigned short* childb = (unsigned short*)(ws + 51200256);      // (NC+1)*128 B
    unsigned short* W1p    = (unsigned short*)(ws + 89600512);      // 73728 B
    unsigned short* W2p    = (unsigned short*)(ws + 89674240);      // 73728 B
    unsigned short* Wv1p   = (unsigned short*)(ws + 89747968);      // 8192 B
    unsigned short* Wv2p   = (unsigned short*)(ws + 89756160);      // 8192 B

    // 1) feat -> bf16, zero the two pad rows
    {
        int n4 = (NP * HID) / 4;                       // 3,200,000
        k_cvt_bf16<<<(n4 + 255) / 256, 256, 0, stream>>>(feat, featb, n4);
        k_zero_pads<<<1, 16, 0, stream>>>(featb + (size_t)NP * HID,
                                          childb + (size_t)NC * HID);
    }
    // 2) pack weights into WMMA B-fragment layout
    {
        int t1 = 576 * 64;
        k_pack_w<<<(t1 + 255) / 256, 256, 0, stream>>>(W1, W1p, 576);
        k_pack_w<<<(t1 + 255) / 256, 256, 0, stream>>>(W2, W2p, 576);
        int t2 = 64 * 64;
        k_pack_w<<<(t2 + 255) / 256, 256, 0, stream>>>(Wv1, Wv1p, 64);
        k_pack_w<<<(t2 + 255) / 256, 256, 0, stream>>>(Wv2, Wv2p, 64);
    }
    // 3) layer 1 quadconv + heads (h kept in bf16 for layer 2)
    {
        int blocks = (NP + ROWS_BLK - 1) / ROWS_BLK;   // 3125
        k_quadconv<<<blocks, 256, LDS_BYTES, stream>>>(
            featb, neigh_d, W1p, b1, Wv1p, Ws1, h1b, split1, value1, NP, NP);
    }
    // 4) teacher-forced gate + parent gather -> child_in (bf16)
    {
        int m8 = NC * 8;
        k_child_gather<<<(m8 + 255) / 256, 256, 0, stream>>>(h1b, parent, gate,
                                                             childb, m8);
    }
    // 5) layer 2 quadconv + heads
    {
        int blocks = (NC + ROWS_BLK - 1) / ROWS_BLK;   // 4688
        k_quadconv<<<blocks, 256, LDS_BYTES, stream>>>(
            childb, neigh_c, W2p, b2, Wv2p, Ws2, nullptr, split2, value2, NC, NC);
    }
}
